// ArcFaceLayer_6897717477494
// MI455X (gfx1250) — matrix operand order repaired and checked by reading, not compiled, not run on hardware
//
#include <hip/hip_runtime.h>
#include <math.h>

// ---------------- types ----------------
typedef __attribute__((ext_vector_type(16))) __bf16 v16bf;
typedef __attribute__((ext_vector_type(8)))  __bf16 bf16x8;
typedef __attribute__((ext_vector_type(8)))  float  v8f;

#define BDIM 512        // batch (M)
#define EDIM 512        // embed (K)
#define CNUM 100000     // classes (N)
#define BN   32         // N cols per block
#define BSTRIDE 528     // LDS K-stride per column (512 + 16 pad -> bank spread)

#define COS_M  0.87758256189037271612f
#define SIN_M  0.47942553860420300538f
#define TH_V  -0.87758256189037271612f
#define MM_V   0.23971276930210150269f

// LDS layout (bytes)
#define OFF_BHI   0
#define OFF_BLO   33792            // 32*528*2
#define OFF_LAB   67584            // + 32*528*2
#define OFF_PSUM  69632            // + 512*4
#define OFF_INVW  70656            // + 256*4
#define LDS_BYTES 70784            // + 32*4

// fp32 -> bf16 round-to-nearest-even
__device__ __forceinline__ __bf16 f2bf(float f) {
    unsigned u = __builtin_bit_cast(unsigned, f);
    unsigned r = (u + 0x7FFFu + ((u >> 16) & 1u)) >> 16;
    unsigned short h = (unsigned short)r;
    return __builtin_bit_cast(__bf16, h);
}
__device__ __forceinline__ float bf2f(__bf16 b) {
    unsigned short h = __builtin_bit_cast(unsigned short, b);
    unsigned u = ((unsigned)h) << 16;
    return __builtin_bit_cast(float, u);
}
__device__ __forceinline__ v16bf mk16(bf16x8 a, bf16x8 b) {
    v16bf r;
#pragma unroll
    for (int i = 0; i < 8; ++i) { r[i] = a[i]; r[i + 8] = b[i]; }
    return r;
}

// ---------------- kernel 1: normalize embeddings, split to bf16 hi/lo ----------------
// one wave per row; 8 waves per block; 64 blocks
__global__ void __launch_bounds__(256)
arcface_prep_a(const float* __restrict__ emb,
               __bf16* __restrict__ Ahi, __bf16* __restrict__ Alo) {
    const int wave = threadIdx.x >> 5;
    const int lane = threadIdx.x & 31;
    const int row  = blockIdx.x * 8 + wave;
    const float* p = emb + (size_t)row * EDIM;

    float vals[16];
    float s = 0.f;
#pragma unroll
    for (int i = 0; i < 16; ++i) {
        float v = p[lane + (i << 5)];
        vals[i] = v;
        s += v * v;
    }
#pragma unroll
    for (int off = 16; off > 0; off >>= 1)
        s += __shfl_xor(s, off, 32);
    const float inv = rsqrtf(fmaxf(s, 1e-12f));

#pragma unroll
    for (int i = 0; i < 16; ++i) {
        float v = vals[i] * inv;
        __bf16 hi = f2bf(v);
        __bf16 lo = f2bf(v - bf2f(hi));
        size_t idx = (size_t)row * EDIM + lane + (i << 5);
        Ahi[idx] = hi;
        Alo[idx] = lo;
    }
}

// ---------------- kernel 2: fused column-norm + bf16x3 WMMA GEMM + ArcFace epilogue ----
// block = 256 threads (8 waves). Each block: full M=512 x 32 columns. 3125 blocks.
__global__ void __launch_bounds__(256)
arcface_gemm(const float* __restrict__ w, const int* __restrict__ labels,
             const __bf16* __restrict__ Ahi, const __bf16* __restrict__ Alo,
             float* __restrict__ out) {
    extern __shared__ __align__(16) char smem[];
    __bf16* sBhi  = (__bf16*)(smem + OFF_BHI);
    __bf16* sBlo  = (__bf16*)(smem + OFF_BLO);
    int*    sLab  = (int*)  (smem + OFF_LAB);
    float*  sPsum = (float*)(smem + OFF_PSUM);
    float*  sInvW = (float*)(smem + OFF_INVW);

    const int tid = threadIdx.x;
    const int n0  = blockIdx.x * BN;

    // ---- stage: read w tile once, accumulate column sum-squares, split to bf16 hi/lo ----
    {
        const int col  = tid & 31;       // 0..31
        const int krow = tid >> 5;       // 0..7
        float ps = 0.f;
#pragma unroll 4
        for (int i = 0; i < 64; ++i) {
            const int k = krow + (i << 3);
            const float v = w[(size_t)k * CNUM + n0 + col];
            ps += v * v;
            __bf16 hi = f2bf(v);
            __bf16 lo = f2bf(v - bf2f(hi));
            sBhi[col * BSTRIDE + k] = hi;
            sBlo[col * BSTRIDE + k] = lo;
        }
        sPsum[tid] = ps;
        sLab[tid]       = labels[tid];
        sLab[tid + 256] = labels[tid + 256];
    }
    __syncthreads();
    if (tid < 32) {
        float s = 0.f;
#pragma unroll
        for (int j = 0; j < 8; ++j) s += sPsum[tid + (j << 5)];
        sInvW[tid] = rsqrtf(fmaxf(s, 1e-12f));
    }
    __syncthreads();

    // ---- WMMA main loop ----
    const int lane   = tid & 31;
    const int wv     = tid >> 5;
    const int mwave  = wv << 6;                    // 64 rows per wave
    const int ln15   = lane & 15;
    const int lhalfA = (lane & 16) ? 8 : 0;        // A fragment K-half select
    const int mhalf  = (lane & 16) ? 8 : 0;        // C/D row-half select
    const int bkoff  = (lane & 16) ? 8 : 0;        // B fragment K-half select

    v8f acc[4][2];
#pragma unroll
    for (int mt = 0; mt < 4; ++mt)
#pragma unroll
        for (int nt = 0; nt < 2; ++nt)
            acc[mt][nt] = (v8f){0.f, 0.f, 0.f, 0.f, 0.f, 0.f, 0.f, 0.f};

#pragma unroll 1
    for (int ks = 0; ks < 16; ++ks) {
        const int k0 = ks << 5;

        // B fragments from LDS (shared by all 8 waves)
        v16bf bh[2], bl[2];
#pragma unroll
        for (int nt = 0; nt < 2; ++nt) {
            const int c = (nt << 4) + ln15;
            const __bf16* pb = sBhi + c * BSTRIDE + k0 + bkoff;
            const __bf16* qb = sBlo + c * BSTRIDE + k0 + bkoff;
            bh[nt] = mk16(*(const bf16x8*)pb, *(const bf16x8*)(pb + 16));
            bl[nt] = mk16(*(const bf16x8*)qb, *(const bf16x8*)(qb + 16));
        }

#pragma unroll
        for (int mt = 0; mt < 4; ++mt) {
            const int row = mwave + (mt << 4) + ln15;
            const __bf16* pa = Ahi + ((size_t)row << 9) + k0 + lhalfA;
            const __bf16* pl = Alo + ((size_t)row << 9) + k0 + lhalfA;
            v16bf ah = mk16(*(const bf16x8*)pa, *(const bf16x8*)(pa + 16));
            v16bf al = mk16(*(const bf16x8*)pl, *(const bf16x8*)(pl + 16));
#pragma unroll
            for (int nt = 0; nt < 2; ++nt) {
                acc[mt][nt] = __builtin_amdgcn_wmma_f32_16x16x32_bf16(
                    false, ah, false, bh[nt], (short)0, acc[mt][nt], false, false);
                acc[mt][nt] = __builtin_amdgcn_wmma_f32_16x16x32_bf16(
                    false, ah, false, bl[nt], (short)0, acc[mt][nt], false, false);
                acc[mt][nt] = __builtin_amdgcn_wmma_f32_16x16x32_bf16(
                    false, al, false, bh[nt], (short)0, acc[mt][nt], false, false);
            }
        }
    }

    // ---- ArcFace epilogue: cos -> margin -> scale, label column only ----
#pragma unroll
    for (int mt = 0; mt < 4; ++mt) {
#pragma unroll
        for (int nt = 0; nt < 2; ++nt) {
            const int nl = (nt << 4) + ln15;
            const int n  = n0 + nl;
            const float invw = sInvW[nl];
            const int mbase = mwave + (mt << 4) + mhalf;
#pragma unroll
            for (int j = 0; j < 8; ++j) {
                const int m = mbase + j;
                float ct = acc[mt][nt][j] * invw;
                ct = fminf(fmaxf(ct, -1.0f + 1e-6f), 1.0f - 1e-6f);
                float st = sqrtf(fminf(fmaxf(1.0f - ct * ct, 1e-6f), 1.0f));
                float cmt = ct * COS_M - st * SIN_M;
                cmt = (ct > TH_V) ? cmt : (ct - MM_V);
                float lg = (sLab[m] == n) ? cmt : ct;
                out[(size_t)m * CNUM + n] = lg * 64.0f;
            }
        }
    }
}

// ---------------- launcher ----------------
extern "C" void kernel_launch(void* const* d_in, const int* in_sizes, int n_in,
                              void* d_out, int out_size, void* d_ws, size_t ws_size,
                              hipStream_t stream) {
    const float* emb    = (const float*)d_in[0];   // [512, 512]
    const int*   labels = (const int*)d_in[1];     // [512]
    const float* w      = (const float*)d_in[2];   // [512, 100000]
    float*       out    = (float*)d_out;           // [512, 100000]

    __bf16* Ahi = (__bf16*)d_ws;                       // 512*512 bf16
    __bf16* Alo = Ahi + (size_t)BDIM * EDIM;           // 512*512 bf16

    arcface_prep_a<<<BDIM / 8, 256, 0, stream>>>(emb, Ahi, Alo);
    arcface_gemm<<<CNUM / BN, 256, LDS_BYTES, stream>>>(w, labels, Ahi, Alo, out);
}
